// Network_78056735638092
// MI455X (gfx1250) — matrix operand order repaired and checked by reading, not compiled, not run on hardware
//
#include <hip/hip_runtime.h>
#include <hip/hip_bf16.h>
#include <math.h>

#define N_NODES 30000
#define N_EDGES 480000
#define DIM     256
#define N_GRAPHS 256
#define NL      3
#define OUTC    10
#define ND      ((size_t)N_NODES * DIM)
#define WMAT    (DIM * DIM)   // 65536

typedef __attribute__((ext_vector_type(16))) _Float16 v16h;
typedef __attribute__((ext_vector_type(8)))  _Float16 v8h;
typedef __attribute__((ext_vector_type(8)))  float    v8f;

// Async global->LDS copy path (CDNA5): use if the toolchain exposes the builtins.
#if defined(__has_builtin)
#if __has_builtin(__builtin_amdgcn_global_load_async_to_lds_b128) && \
    __has_builtin(__builtin_amdgcn_s_wait_asynccnt)
#define USE_ASYNC_LDS 1
#endif
#endif
#ifndef USE_ASYNC_LDS
#define USE_ASYNC_LDS 0
#endif

// Builtin takes int4* in AS1 (global) / AS3 (LDS): clang prints AS1 as "__device__".
typedef __attribute__((__vector_size__(4 * sizeof(int)))) int vint4;
typedef __attribute__((address_space(1))) vint4* as1_vint4p;
typedef __attribute__((address_space(3))) vint4* as3_vint4p;

enum { GF_ACC = 1, GF_RELU_F16 = 2, GF_A_F16 = 4, GF_A_ADD = 8, GF_ROWSCALE = 16 };

// ---------------- small utility kernels ----------------

__global__ __launch_bounds__(256) void k_zero_f32(float* p, int n) {
  int i = blockIdx.x * 256 + threadIdx.x;
  if (i < n) p[i] = 0.0f;
}

// Convert f32 weights [nmat][256][256] (K,N row-major) into WMMA-B-fragment order:
// Wt[mat][k>>4][n][k&15] -> per-lane fragment = 16 contiguous halves (32B), lanes coalesced.
__global__ __launch_bounds__(256) void k_cvt_wt(const float* __restrict__ in,
                                                _Float16* __restrict__ out, int n) {
  int i = blockIdx.x * 256 + threadIdx.x;
  if (i >= n) return;
  int m = i >> 16;
  int r = i & 65535;
  int k = r >> 8, c = r & 255;
  out[(size_t)m * WMAT + ((((k >> 4) << 8) + c) << 4) + (k & 15)] = (_Float16)in[i];
}

// scalar mixing weights: sc[0..11]=softmax(na_w) rows, sc[12..23]=softmax(fu_w) rows,
// sc[24..39]=softmax(se_w[i][j])[1]
__global__ void k_prep_scalars(const float* __restrict__ na_w,
                               const float* __restrict__ fu_w,
                               const float* __restrict__ se_w,
                               float* __restrict__ sc) {
  if (threadIdx.x != 0 || blockIdx.x != 0) return;
  for (int i = 0; i < NL; ++i) {            // na_w [3][4]
    const float* r = na_w + i * 4;
    float m = fmaxf(fmaxf(r[0], r[1]), fmaxf(r[2], r[3]));
    float e[4], s = 0.f;
    for (int j = 0; j < 4; ++j) { e[j] = __expf(r[j] - m); s += e[j]; }
    for (int j = 0; j < 4; ++j) sc[i * 4 + j] = e[j] / s;
  }
  for (int i = 0; i < NL + 1; ++i) {        // fu_w [4][3]
    const float* r = fu_w + i * 3;
    float m = fmaxf(r[0], fmaxf(r[1], r[2]));
    float e[3], s = 0.f;
    for (int j = 0; j < 3; ++j) { e[j] = __expf(r[j] - m); s += e[j]; }
    for (int j = 0; j < 3; ++j) sc[12 + i * 3 + j] = e[j] / s;
  }
  for (int i = 0; i < NL + 1; ++i)          // se_w [4][4][2] -> softmax[..,1]
    for (int j = 0; j <= i; ++j) {
      float a = se_w[(i * 4 + j) * 2 + 0], b = se_w[(i * 4 + j) * 2 + 1];
      float m = fmaxf(a, b);
      float ea = __expf(a - m), eb = __expf(b - m);
      sc[24 + i * 4 + j] = eb / (ea + eb);
    }
}

__global__ __launch_bounds__(256) void k_deg(const int* __restrict__ dst, float* deg) {
  int e = blockIdx.x * 256 + threadIdx.x;
  if (e < N_EDGES) atomicAdd(&deg[dst[e]], 1.0f);
}

__global__ __launch_bounds__(256) void k_node_prep(const float* __restrict__ deg,
                                                   float* dis, float* invdeg) {
  int n = blockIdx.x * 256 + threadIdx.x;
  if (n < N_NODES) {
    float d = deg[n];
    dis[n] = rsqrtf(d + 1.0f);
    invdeg[n] = 1.0f / fmaxf(d, 1.0f);
  }
}

__global__ __launch_bounds__(256) void k_ecoef(const int* __restrict__ src,
                                               const int* __restrict__ dst,
                                               const float* __restrict__ dis,
                                               float* ecoef) {
  int e = blockIdx.x * 256 + threadIdx.x;
  if (e < N_EDGES) ecoef[e] = dis[src[e]] * dis[dst[e]];
}

// ---------------- SE/FU fuse + BN ----------------

__global__ __launch_bounds__(256) void k_fused(const float* __restrict__ emb,
                                               const float* __restrict__ h1,
                                               const float* __restrict__ h2,
                                               const float* __restrict__ h3,
                                               const float* __restrict__ sc,
                                               int layer, float* __restrict__ x2,
                                               float* colsum, float* colsumsq) {
  const int d = threadIdx.x;                 // 256 columns
  const int rbeg = blockIdx.x * 128;
  const float f0 = sc[12 + layer * 3 + 0];
  const float f1 = sc[12 + layer * 3 + 1];
  const float f2 = sc[12 + layer * 3 + 2];
  const float s0 = sc[24 + layer * 4 + 0];
  const float s1 = (layer >= 1) ? sc[24 + layer * 4 + 1] : 0.f;
  const float s2 = (layer >= 2) ? sc[24 + layer * 4 + 2] : 0.f;
  const float s3 = (layer >= 3) ? sc[24 + layer * 4 + 3] : 0.f;
  const float invc = 1.0f / (float)(layer + 1);
  const float e0 = s0 * emb[d];
  float lsum = 0.f, lsq = 0.f;
  for (int r = 0; r < 128; ++r) {
    int n = rbeg + r;
    if (n >= N_NODES) break;
    size_t off = (size_t)n * DIM + d;
    float sum = e0, mx = e0;
    if (layer >= 1) { float v = s1 * h1[off]; sum += v; mx = fmaxf(mx, v); }
    if (layer >= 2) { float v = s2 * h2[off]; sum += v; mx = fmaxf(mx, v); }
    if (layer >= 3) { float v = s3 * h3[off]; sum += v; mx = fmaxf(mx, v); }
    float fused = f0 * sum + f1 * (sum * invc) + f2 * mx;
    x2[off] = fused;
    lsum += fused; lsq += fused * fused;
  }
  atomicAdd(&colsum[d], lsum);
  atomicAdd(&colsumsq[d], lsq);
}

__global__ void k_bn_finalize(const float* colsum, const float* colsumsq,
                              const float* gamma, const float* beta, int layer,
                              float* scale, float* shift) {
  int d = threadIdx.x;
  const float invN = 1.0f / (float)N_NODES;
  float m = colsum[d] * invN;
  float var = colsumsq[d] * invN - m * m;
  float s = gamma[layer * DIM + d] * rsqrtf(var + 1e-5f);
  scale[d] = s;
  shift[d] = beta[layer * DIM + d] - m * s;
}

__global__ __launch_bounds__(256) void k_bn_elu(float* __restrict__ x2,
                                                const float* __restrict__ scale,
                                                const float* __restrict__ shift) {
  size_t i = (size_t)blockIdx.x * 256 + threadIdx.x;
  if (i >= ND) return;
  int d = (int)(i & (DIM - 1));
  float v = scale[d] * x2[i] + shift[d];
  x2[i] = (v > 0.f) ? v : expm1f(v);
}

// ---------------- message passing ----------------

__global__ __launch_bounds__(256) void k_edge_agg(const int* __restrict__ src,
                                                  const int* __restrict__ dst,
                                                  const float* __restrict__ x,
                                                  float* __restrict__ A) {
  int tid = blockIdx.x * 256 + threadIdx.x;
  int e = tid >> 6, q = (tid & 63) << 2;
  if (e >= N_EDGES) return;
  const float4 v = *(const float4*)(x + (size_t)src[e] * DIM + q);
  float* o = A + (size_t)dst[e] * DIM + q;
  atomicAdd(o + 0, v.x); atomicAdd(o + 1, v.y);
  atomicAdd(o + 2, v.z); atomicAdd(o + 3, v.w);
}

__global__ __launch_bounds__(256) void k_gcn_self(const float* __restrict__ hw,
                                                  const float* __restrict__ dis,
                                                  const float* __restrict__ b,
                                                  const float* __restrict__ sc,
                                                  int layer, float* __restrict__ X) {
  size_t i = (size_t)blockIdx.x * 256 + threadIdx.x;
  if (i >= ND) return;
  int n = (int)(i >> 8), d = (int)(i & 255);
  float a0 = sc[layer * 4 + 0];
  float di = dis[n];
  X[i] = a0 * (di * di * hw[i] + b[d]);
}

__global__ __launch_bounds__(256) void k_gcn_edge(const int* __restrict__ src,
                                                  const int* __restrict__ dst,
                                                  const float* __restrict__ hw,
                                                  const float* __restrict__ ecoef,
                                                  const float* __restrict__ sc,
                                                  int layer, float* __restrict__ X) {
  int tid = blockIdx.x * 256 + threadIdx.x;
  int e = tid >> 6, q = (tid & 63) << 2;
  if (e >= N_EDGES) return;
  float c = sc[layer * 4 + 0] * ecoef[e];
  const float4 v = *(const float4*)(hw + (size_t)src[e] * DIM + q);
  float* o = X + (size_t)dst[e] * DIM + q;
  atomicAdd(o + 0, c * v.x); atomicAdd(o + 1, c * v.y);
  atomicAdd(o + 2, c * v.z); atomicAdd(o + 3, c * v.w);
}

__global__ __launch_bounds__(256) void k_pool(const int* __restrict__ batch,
                                              const float* __restrict__ x,
                                              float* __restrict__ pooled) {
  int tid = blockIdx.x * 256 + threadIdx.x;
  int n = tid >> 6, q = (tid & 63) << 2;
  if (n >= N_NODES) return;
  int g = batch[n];
  const float4 v = *(const float4*)(x + (size_t)n * DIM + q);
  float* o = pooled + (size_t)g * DIM + q;
  atomicAdd(o + 0, v.x); atomicAdd(o + 1, v.y);
  atomicAdd(o + 2, v.z); atomicAdd(o + 3, v.w);
}

__global__ __launch_bounds__(256) void k_classifier(const float* __restrict__ pooled,
                                                    const float* __restrict__ Wc,
                                                    const float* __restrict__ bc,
                                                    float* __restrict__ out) {
  int idx = blockIdx.x * 256 + threadIdx.x;
  if (idx >= N_GRAPHS * OUTC) return;
  int g = idx / OUTC, o = idx % OUTC;
  float s = bc[o];
  for (int d = 0; d < DIM; ++d) s += pooled[(size_t)g * DIM + d] * Wc[d * OUTC + o];
  out[idx] = s;
}

// ---------------- WMMA GEMM:  C (+)= alpha * (A[N,256] @ W[256,256] + bias) ----------------
// Block = 256 threads = 8 waves; block tile = 256 rows x 64 cols.
// The whole 256x64 f16 weight tile (32KB) is staged into LDS once per block —
// via GLOBAL_LOAD_ASYNC_TO_LDS_B128 (ASYNCcnt) when the builtin is available,
// else coalesced b128 loads + ds stores — then one barrier; the K-loop reads B
// via ds_load_b128. Each wave computes a 32x64 tile: 2x4 accumulators, 8
// v_wmma_f32_16x16x32_f16 per K-step fed by just 2 A fragments.

template <int FLAGS>
__global__ __launch_bounds__(256) void k_gemm_wmma(
    const float*    __restrict__ A0,
    const float*    __restrict__ A1,
    const _Float16* __restrict__ A16,
    const float*    __restrict__ rowscale,
    const _Float16* __restrict__ Wt,      // shuffled: [k>>4][col][k&15]
    const float*    __restrict__ bias,
    const float*    __restrict__ alpha_p,
    float*          __restrict__ C,
    _Float16*       __restrict__ Oh,
    int nrows) {
  __shared__ _Float16 Bs[16 * 64 * 16];   // [kblk][colLocal][j], 32KB

  const int t  = threadIdx.x;
  const int cb = blockIdx.y * 64;

  // cooperative stage of the block's weight tile: 2048 x 16B units
#pragma unroll
  for (int c = 0; c < 8; ++c) {
    const int u = c * 256 + t;
    const int h8   = (u & 1) << 3;        // which 8-half chunk
    const int col  = (u >> 1) & 63;
    const int kblk = u >> 7;
    const size_t s = ((size_t)((kblk << 8) + cb + col) << 4) + h8;
    const int    d = (((kblk << 6) + col) << 4) + h8;
#if USE_ASYNC_LDS
    __builtin_amdgcn_global_load_async_to_lds_b128(
        (as1_vint4p)(Wt + s), (as3_vint4p)(Bs + d), 0, 0);
#else
    *(v8h*)(Bs + d) = *(const v8h*)(Wt + s);
#endif
  }
#if USE_ASYNC_LDS
  __builtin_amdgcn_s_wait_asynccnt(0);
#endif
  __syncthreads();

  const int lane  = t & 31;
  const int wave  = t >> 5;
  const int r0    = blockIdx.x * 256 + wave * 32;
  const int lrow  = lane & 15;
  const int khalf = lane >> 4;

  const int rowA[2] = { r0 + lrow, r0 + 16 + lrow };
  float rs[2] = { 1.0f, 1.0f };
  if (FLAGS & GF_ROWSCALE) {
#pragma unroll
    for (int mi = 0; mi < 2; ++mi)
      rs[mi] = (rowA[mi] < nrows) ? rowscale[rowA[mi]] : 0.0f;
  }

  v8f acc[2][4] = {};

  for (int kk = 0; kk < DIM; kk += 32) {
    v16h a[2], b[4];
    const int k0 = kk + khalf * 16;       // elements 0..7
    const int k1 = kk + khalf * 16 + 8;   // elements 8..15
#pragma unroll
    for (int mi = 0; mi < 2; ++mi) {
      const int row = rowA[mi];
      if (row < nrows) {
        const size_t rb = (size_t)row * DIM;
        if (FLAGS & GF_A_F16) {
          v8h h0 = *(const v8h*)(A16 + rb + k0);
          v8h h1 = *(const v8h*)(A16 + rb + k1);
          a[mi] = __builtin_shufflevector(h0, h1, 0, 1, 2, 3, 4, 5, 6, 7,
                                          8, 9, 10, 11, 12, 13, 14, 15);
        } else {
          v8f f0 = *(const v8f*)(A0 + rb + k0);
          v8f f1 = *(const v8f*)(A0 + rb + k1);
          if (FLAGS & GF_A_ADD) {
            f0 += *(const v8f*)(A1 + rb + k0);
            f1 += *(const v8f*)(A1 + rb + k1);
          }
          if (FLAGS & GF_ROWSCALE) { f0 *= rs[mi]; f1 *= rs[mi]; }
          v8h h0 = __builtin_convertvector(f0, v8h);
          v8h h1 = __builtin_convertvector(f1, v8h);
          a[mi] = __builtin_shufflevector(h0, h1, 0, 1, 2, 3, 4, 5, 6, 7,
                                          8, 9, 10, 11, 12, 13, 14, 15);
        }
      } else {
        v16h z = {};
        a[mi] = z;
      }
    }
#pragma unroll
    for (int ni = 0; ni < 4; ++ni) {
      const int colL = ni * 16 + lrow;
      const int off = ((((kk >> 4) + khalf) << 6) + colL) << 4;
      v8h b0 = *(const v8h*)(Bs + off);
      v8h b1 = *(const v8h*)(Bs + off + 8);
      b[ni] = __builtin_shufflevector(b0, b1, 0, 1, 2, 3, 4, 5, 6, 7,
                                      8, 9, 10, 11, 12, 13, 14, 15);
    }
#pragma unroll
    for (int mi = 0; mi < 2; ++mi)
#pragma unroll
      for (int ni = 0; ni < 4; ++ni)
        acc[mi][ni] = __builtin_amdgcn_wmma_f32_16x16x32_f16(
            false, a[mi], false, b[ni], (short)0, acc[mi][ni], false, false);
  }

  const float alpha = alpha_p ? *alpha_p : 1.0f;
#pragma unroll
  for (int mi = 0; mi < 2; ++mi)
#pragma unroll
    for (int ni = 0; ni < 4; ++ni) {
      const int col = cb + ni * 16 + lrow;
      const float bv = bias ? bias[col] : 0.0f;
#pragma unroll
      for (int v = 0; v < 8; ++v) {
        const int row = r0 + mi * 16 + v + 8 * khalf;   // C/D layout: VGPR v, lane half
        if (row >= nrows) continue;
        const float val = acc[mi][ni][v] + bv;
        const size_t idx = (size_t)row * DIM + col;
        if (FLAGS & GF_RELU_F16)      Oh[idx] = (_Float16)fmaxf(val, 0.0f);
        else if (FLAGS & GF_ACC)      C[idx] += alpha * val;
        else                          C[idx]  = alpha * val;
      }
    }
}

// ---------------- host orchestration ----------------

extern "C" void kernel_launch(void* const* d_in, const int* in_sizes, int n_in,
                              void* d_out, int out_size, void* d_ws, size_t ws_size,
                              hipStream_t stream) {
  const float* emb    = (const float*)d_in[0];
  const float* na_w   = (const float*)d_in[1];
  const float* fu_w   = (const float*)d_in[2];
  const float* se_w   = (const float*)d_in[3];
  const float* gcn_W  = (const float*)d_in[4];
  const float* gcn_b  = (const float*)d_in[5];
  const float* sage_Wl= (const float*)d_in[6];
  const float* sage_Wr= (const float*)d_in[7];
  const float* sage_b = (const float*)d_in[8];
  const float* gin_W1 = (const float*)d_in[9];
  const float* gin_b1 = (const float*)d_in[10];
  const float* gin_W2 = (const float*)d_in[11];
  const float* gin_b2 = (const float*)d_in[12];
  const float* gc_Wl  = (const float*)d_in[13];
  const float* gc_Wr  = (const float*)d_in[14];
  const float* gc_b   = (const float*)d_in[15];
  const float* bn_g   = (const float*)d_in[16];
  const float* bn_b   = (const float*)d_in[17];
  const float* cls_W  = (const float*)d_in[18];
  const float* cls_b  = (const float*)d_in[19];
  const int*   ei     = (const int*)d_in[21];
  const int*   batch  = (const int*)d_in[22];
  const int* src = ei;
  const int* dst = ei + N_EDGES;

  char* p = (char*)d_ws;
  auto take = [&](size_t bytes) -> char* {
    char* r = p;
    p += (bytes + 255) & ~(size_t)255;
    return r;
  };
  float* sc       = (float*)take(64 * sizeof(float));
  float* deg      = (float*)take(N_NODES * 4);
  float* dis      = (float*)take(N_NODES * 4);
  float* invdeg   = (float*)take(N_NODES * 4);
  float* ecoef    = (float*)take(N_EDGES * 4);
  float* colsum   = (float*)take(DIM * 4);
  float* colsumsq = (float*)take(DIM * 4);
  float* bnscale  = (float*)take(DIM * 4);
  float* bnshift  = (float*)take(DIM * 4);
  float* hbuf     = (float*)take(3 * ND * 4);   // h1,h2,h3 (h0 == emb broadcast, implicit)
  float* x2       = (float*)take(ND * 4);
  float* Ab       = (float*)take(ND * 4);
  float* hwb      = (float*)take(ND * 4);
  _Float16* gmid  = (_Float16*)take(ND * 2);
  _Float16* whAll = (_Float16*)take((size_t)7 * NL * WMAT * 2);
  float* pooled   = (float*)take((size_t)N_GRAPHS * DIM * 4);

  _Float16* wt_gcn = whAll + 0 * NL * WMAT;
  _Float16* wt_sWl = whAll + 1 * NL * WMAT;
  _Float16* wt_sWr = whAll + 2 * NL * WMAT;
  _Float16* wt_g1  = whAll + 3 * NL * WMAT;
  _Float16* wt_g2  = whAll + 4 * NL * WMAT;
  _Float16* wt_cl  = whAll + 5 * NL * WMAT;
  _Float16* wt_cr  = whAll + 6 * NL * WMAT;

  const int BLK = 256;
  const int gNodes  = (N_NODES + BLK - 1) / BLK;
  const int gEdges  = (N_EDGES + BLK - 1) / BLK;
  const int gND     = (int)((ND + BLK - 1) / BLK);
  const int gEdgeV  = (int)(((size_t)N_EDGES * 64 + BLK - 1) / BLK);
  const int gNodeV  = (int)(((size_t)N_NODES * 64 + BLK - 1) / BLK);
  const int gWcvt   = (NL * WMAT + BLK - 1) / BLK;

  // static prep
  k_prep_scalars<<<1, 1, 0, stream>>>(na_w, fu_w, se_w, sc);
  k_zero_f32<<<gNodes, BLK, 0, stream>>>(deg, N_NODES);
  k_deg<<<gEdges, BLK, 0, stream>>>(dst, deg);
  k_node_prep<<<gNodes, BLK, 0, stream>>>(deg, dis, invdeg);
  k_ecoef<<<gEdges, BLK, 0, stream>>>(src, dst, dis, ecoef);
  k_cvt_wt<<<gWcvt, BLK, 0, stream>>>(gcn_W,   wt_gcn, NL * WMAT);
  k_cvt_wt<<<gWcvt, BLK, 0, stream>>>(sage_Wl, wt_sWl, NL * WMAT);
  k_cvt_wt<<<gWcvt, BLK, 0, stream>>>(sage_Wr, wt_sWr, NL * WMAT);
  k_cvt_wt<<<gWcvt, BLK, 0, stream>>>(gin_W1,  wt_g1,  NL * WMAT);
  k_cvt_wt<<<gWcvt, BLK, 0, stream>>>(gin_W2,  wt_g2,  NL * WMAT);
  k_cvt_wt<<<gWcvt, BLK, 0, stream>>>(gc_Wl,   wt_cl,  NL * WMAT);
  k_cvt_wt<<<gWcvt, BLK, 0, stream>>>(gc_Wr,   wt_cr,  NL * WMAT);

  const dim3 gGemm((N_NODES + 255) / 256, DIM / 64);
  const int gFuse = (N_NODES + 127) / 128;
  float* h1 = hbuf, *h2 = hbuf + ND, *h3 = hbuf + 2 * ND;

  for (int i = 0; i < NL; ++i) {
    // SE/FU fuse + BN + ELU -> x2
    k_zero_f32<<<1, BLK, 0, stream>>>(colsum, DIM);
    k_zero_f32<<<1, BLK, 0, stream>>>(colsumsq, DIM);
    k_fused<<<gFuse, BLK, 0, stream>>>(emb, h1, h2, h3, sc, i, x2, colsum, colsumsq);
    k_bn_finalize<<<1, DIM, 0, stream>>>(colsum, colsumsq, bn_g, bn_b, i, bnscale, bnshift);
    k_bn_elu<<<gND, BLK, 0, stream>>>(x2, bnscale, bnshift);

    // shared neighbor sum A = agg_sum(x2)
    k_zero_f32<<<gND, BLK, 0, stream>>>(Ab, (int)ND);
    k_edge_agg<<<gEdgeV, BLK, 0, stream>>>(src, dst, x2, Ab);

    float* X = hbuf + (size_t)i * ND;  // h_{i+1}

    // GCN: hw = x2 @ W, then self term + normalized scatter (scaled by nw0)
    k_gemm_wmma<0><<<gGemm, BLK, 0, stream>>>(
        x2, nullptr, nullptr, nullptr, wt_gcn + (size_t)i * WMAT,
        nullptr, nullptr, hwb, nullptr, N_NODES);
    k_gcn_self<<<gND, BLK, 0, stream>>>(hwb, dis, gcn_b + i * DIM, sc, i, X);
    k_gcn_edge<<<gEdgeV, BLK, 0, stream>>>(src, dst, hwb, ecoef, sc, i, X);

    // SAGE: X += nw1 * ((A/deg) @ Wl + b) + nw1 * (x2 @ Wr)
    k_gemm_wmma<GF_ACC | GF_ROWSCALE><<<gGemm, BLK, 0, stream>>>(
        Ab, nullptr, nullptr, invdeg, wt_sWl + (size_t)i * WMAT,
        sage_b + i * DIM, sc + i * 4 + 1, X, nullptr, N_NODES);
    k_gemm_wmma<GF_ACC><<<gGemm, BLK, 0, stream>>>(
        x2, nullptr, nullptr, nullptr, wt_sWr + (size_t)i * WMAT,
        nullptr, sc + i * 4 + 1, X, nullptr, N_NODES);

    // GIN: gmid = relu((x2 + A) @ W1 + b1); X += nw2 * (gmid @ W2 + b2)
    k_gemm_wmma<GF_A_ADD | GF_RELU_F16><<<gGemm, BLK, 0, stream>>>(
        x2, Ab, nullptr, nullptr, wt_g1 + (size_t)i * WMAT,
        gin_b1 + i * DIM, nullptr, nullptr, gmid, N_NODES);
    k_gemm_wmma<GF_ACC | GF_A_F16><<<gGemm, BLK, 0, stream>>>(
        nullptr, nullptr, gmid, nullptr, wt_g2 + (size_t)i * WMAT,
        gin_b2 + i * DIM, sc + i * 4 + 2, X, nullptr, N_NODES);

    // GraphConv: X += nw3 * (A @ Wl + b) + nw3 * (x2 @ Wr)
    k_gemm_wmma<GF_ACC><<<gGemm, BLK, 0, stream>>>(
        Ab, nullptr, nullptr, nullptr, wt_cl + (size_t)i * WMAT,
        gc_b + i * DIM, sc + i * 4 + 3, X, nullptr, N_NODES);
    k_gemm_wmma<GF_ACC><<<gGemm, BLK, 0, stream>>>(
        x2, nullptr, nullptr, nullptr, wt_cr + (size_t)i * WMAT,
        nullptr, sc + i * 4 + 3, X, nullptr, N_NODES);
  }

  // final: fuse + BN + ELU -> pool -> classify
  k_zero_f32<<<1, BLK, 0, stream>>>(colsum, DIM);
  k_zero_f32<<<1, BLK, 0, stream>>>(colsumsq, DIM);
  k_fused<<<gFuse, BLK, 0, stream>>>(emb, h1, h2, h3, sc, NL, x2, colsum, colsumsq);
  k_bn_finalize<<<1, DIM, 0, stream>>>(colsum, colsumsq, bn_g, bn_b, NL, bnscale, bnshift);
  k_bn_elu<<<gND, BLK, 0, stream>>>(x2, bnscale, bnshift);

  k_zero_f32<<<(N_GRAPHS * DIM + BLK - 1) / BLK, BLK, 0, stream>>>(pooled, N_GRAPHS * DIM);
  k_pool<<<gNodeV, BLK, 0, stream>>>(batch, x2, pooled);
  k_classifier<<<(N_GRAPHS * OUTC + BLK - 1) / BLK, BLK, 0, stream>>>(
      pooled, cls_W, cls_b, (float*)d_out);
}
